// W4A16LinearStatic_78331613545153
// MI455X (gfx1250) — compile-verified, matrix operand order checked by reading, not run on hardware
//
#include <hip/hip_runtime.h>
#include <stdint.h>

// ---------------------------------------------------------------------------
// W4A32 linear for MI455X (gfx1250): Y = X @ dequant(Wp) + bias
//   X  : [8192, 4096]  f32
//   Wp : [4096, 5504]  int32 (one byte each = two signed int4, lo nibble = even col)
//   scale/offset/bias : [11008] f32,  w = (q + offset) * scale
//   Y  : [8192, 11008] f32
//
// Strategy (compute-bound, AI ~1260 flop/byte -> matrix pipe sets the ceiling):
//   Pass 1: X  -> bf16            (one shot, ~9 us HBM-bound @ 23.3 TB/s)
//   Pass 2: Wp -> dequant bf16^T  [N][K] (one shot, ~9 us HBM-bound)
//   Pass 3: pure bf16 WMMA GEMM, tiles staged with CDNA5 async global->LDS
//           copies (ASYNCcnt), LDS double-buffered, zero staging VALU.
// Fallback: fused dequant-in-loop kernel if d_ws is too small for 157 MB.
// ---------------------------------------------------------------------------

constexpr int M = 8192;
constexpr int K = 4096;
constexpr int N = 11008;
constexpr int NPK = N / 2;      // packed columns

constexpr int BM = 128;
constexpr int BN = 128;
constexpr int BK = 64;

typedef __attribute__((ext_vector_type(16))) __bf16    v16bf;
typedef __attribute__((ext_vector_type(8)))  float     v8f;
typedef __attribute__((ext_vector_type(8)))  uint16_t  u16x8;

union FragAB {
    v16bf v;
    u16x8 h[2];
};

__device__ __forceinline__ uint16_t f32_to_bf16(float f) {
    uint32_t u = __float_as_uint(f);
    u += 0x7fffu + ((u >> 16) & 1u);          // round-to-nearest-even
    return (uint16_t)(u >> 16);
}

__device__ __forceinline__ uint32_t pack_bf16x2(float a, float b) {
    return (uint32_t)f32_to_bf16(a) | ((uint32_t)f32_to_bf16(b) << 16);
}

// Low 32 bits of a flat shared-aperture pointer == LDS byte address.
__device__ __forceinline__ uint32_t lds_addr_of(const void* p) {
    return (uint32_t)(uintptr_t)p;
}

// CDNA5 async copy: 16 B per lane from global memory straight into LDS, same offset on both.
// IOFFSET applies to BOTH sides (ISA 10.x async pseudocode), so one base pair
// serves all four 16B chunks -> no per-chunk address VALU.
template <int OFF>
__device__ __forceinline__ void async_b128(uint32_t lds, const uint16_t* g) {
    asm volatile("global_load_async_to_lds_b128 %0, %1, off offset:%2"
                 :: "v"(lds), "v"(g), "n"(OFF) : "memory");
}
// 64 B per lane (one 128x64-bf16 tile slice: row r, 32 cols).
__device__ __forceinline__ void async_tile_slice(uint32_t lds, const uint16_t* g) {
    async_b128<0>(lds, g);
    async_b128<16>(lds, g);
    async_b128<32>(lds, g);
    async_b128<48>(lds, g);
}

__device__ __forceinline__ void wait_asynccnt0() {
    asm volatile("s_wait_asynccnt 0x0" ::: "memory");
}

// ---------------------------------------------------------------------------
// Pass 1: X (f32) -> Xb (bf16), 8 elements/thread
// ---------------------------------------------------------------------------
__global__ __launch_bounds__(256)
void cvt_x_kernel(const float* __restrict__ X, uint16_t* __restrict__ Xb) {
    const size_t i = ((size_t)blockIdx.x * 256 + threadIdx.x) * 8;
    float4 a = ((const float4*)(X + i))[0];
    float4 b = ((const float4*)(X + i))[1];
    uint4 o = make_uint4(pack_bf16x2(a.x, a.y), pack_bf16x2(a.z, a.w),
                         pack_bf16x2(b.x, b.y), pack_bf16x2(b.z, b.w));
    *(uint4*)(Xb + i) = o;
}

// ---------------------------------------------------------------------------
// Pass 2: Wp (packed int4) -> Wt (bf16, transposed [N][K], dequantized)
// grid (NPK/64, K/64), 256 threads; thread: 1 packed col x 16 k
// ---------------------------------------------------------------------------
__global__ __launch_bounds__(256)
void dequant_wt_kernel(const int*   __restrict__ Wp,
                       const float* __restrict__ scale,
                       const float* __restrict__ offs,
                       uint16_t*    __restrict__ Wt) {
    const int tid = threadIdx.x;
    const int j  = blockIdx.x * 64 + (tid & 63);        // packed col
    const int k0 = blockIdx.y * 64 + (tid >> 6) * 16;   // k start

    const float sc0 = scale[2 * j + 0], sc1 = scale[2 * j + 1];
    const float of0 = offs [2 * j + 0], of1 = offs [2 * j + 1];

    uint32_t lo[8], hi[8];
    #pragma unroll
    for (int i = 0; i < 8; ++i) {
        int p0 = Wp[(size_t)(k0 + 2 * i + 0) * NPK + j];
        int p1 = Wp[(size_t)(k0 + 2 * i + 1) * NPK + j];
        float l0 = (float)(((p0 & 15) ^ 8) - 8);
        float h0 = (float)((((p0 >> 4) & 15) ^ 8) - 8);
        float l1 = (float)(((p1 & 15) ^ 8) - 8);
        float h1 = (float)((((p1 >> 4) & 15) ^ 8) - 8);
        lo[i] = pack_bf16x2((l0 + of0) * sc0, (l1 + of0) * sc0);
        hi[i] = pack_bf16x2((h0 + of1) * sc1, (h1 + of1) * sc1);
    }
    uint32_t* d0 = (uint32_t*)&Wt[(size_t)(2 * j + 0) * K + k0];
    uint32_t* d1 = (uint32_t*)&Wt[(size_t)(2 * j + 1) * K + k0];
    #pragma unroll
    for (int i = 0; i < 8; ++i) { d0[i] = lo[i]; d1[i] = hi[i]; }
}

// ---------------------------------------------------------------------------
// Pass 3: pure bf16 WMMA GEMM. A=[M][K] bf16, B=Wt=[N][K] bf16 (both row-major
// along K), async-staged into double-buffered LDS (buffers 16 KB apart ->
// branch-free buffer select by OR-ing ((kt+1)&1)<<14 into the LDS address).
// ---------------------------------------------------------------------------
__global__ __launch_bounds__(256)
void bf16_wmma_gemm(const uint16_t* __restrict__ Ab,
                    const uint16_t* __restrict__ Bt,
                    const float*    __restrict__ bias,
                    float*          __restrict__ Y) {
    __shared__ uint16_t Als[2][BM][BK];   // 2 x 16 KB
    __shared__ uint16_t Bls[2][BN][BK];   // 2 x 16 KB

    const int tid  = threadIdx.x;
    const int wave = tid >> 5;
    const int lane = tid & 31;
    const int half = lane >> 4;
    const int lr   = lane & 15;

    const int m0 = blockIdx.y * BM;
    const int n0 = blockIdx.x * BN;

    const int wm = wave & 3;           // 4 waves along M -> 32 rows
    const int wn = wave >> 2;          // 2 waves along N -> 64 cols

    // copy assignment: 2 threads per tile row, 32 bf16 (64 B) each
    const int r = tid >> 1;
    const int c = (tid & 1) * 32;

    const uint16_t* Ag = Ab + (size_t)(m0 + r) * K + c;
    const uint16_t* Bg = Bt + (size_t)(n0 + r) * K + c;

    const uint32_t aLb = lds_addr_of(&Als[0][r][c]);   // buffer-0 base
    const uint32_t bLb = lds_addr_of(&Bls[0][r][c]);

    v8f acc[2][4];
    #pragma unroll
    for (int i = 0; i < 2; ++i)
        #pragma unroll
        for (int j = 0; j < 4; ++j)
            acc[i][j] = v8f{};

    // prologue: tile 0 -> buffer 0
    async_tile_slice(aLb, Ag);
    async_tile_slice(bLb, Bg);

    for (int kt = 0; kt < K / BK; ++kt) {
        const int buf = kt & 1;

        wait_asynccnt0();      // this wave's copies of tile kt landed in LDS
        __syncthreads();       // everyone's copies visible; prev buf free

        if (kt + 1 < K / BK) { // stage tile kt+1 into the other buffer
            const uint32_t bo = (uint32_t)((kt + 1) & 1) << 14;  // 16 KB stride
            async_tile_slice(aLb + bo, Ag + (size_t)(kt + 1) * BK);
            async_tile_slice(bLb + bo, Bg + (size_t)(kt + 1) * BK);
        }

        // ---- compute: 2 k-steps of 32, 2x4 WMMA tiles per wave ----
        #pragma unroll
        for (int ks = 0; ks < 2; ++ks) {
            const int k0 = ks * 32 + half * 8;   // lanes 0-15: K 0..7 & 16..23
            FragAB a[2], b[4];
            #pragma unroll
            for (int mt = 0; mt < 2; ++mt) {
                const uint16_t* p = &Als[buf][wm * 32 + mt * 16 + lr][k0];
                a[mt].h[0] = *(const u16x8*)(p);
                a[mt].h[1] = *(const u16x8*)(p + 16);
            }
            #pragma unroll
            for (int nt = 0; nt < 4; ++nt) {
                const uint16_t* p = &Bls[buf][wn * 64 + nt * 16 + lr][k0];
                b[nt].h[0] = *(const u16x8*)(p);
                b[nt].h[1] = *(const u16x8*)(p + 16);
            }
            #pragma unroll
            for (int mt = 0; mt < 2; ++mt)
                #pragma unroll
                for (int nt = 0; nt < 4; ++nt)
                    acc[mt][nt] = __builtin_amdgcn_wmma_f32_16x16x32_bf16(
                        false, a[mt].v, false, b[nt].v,
                        (short)0, acc[mt][nt], false, false);
        }
    }

    // ---- epilogue: bias + store (C/D layout: vgpr i -> row base+i+8*half) ----
    #pragma unroll
    for (int nt = 0; nt < 4; ++nt) {
        const int col  = n0 + wn * 64 + nt * 16 + lr;
        const float bc = bias[col];
        #pragma unroll
        for (int mt = 0; mt < 2; ++mt) {
            const int rbase = m0 + wm * 32 + mt * 16 + half * 8;
            #pragma unroll
            for (int i = 0; i < 8; ++i)
                Y[(size_t)(rbase + i) * N + col] = acc[mt][nt][i] + bc;
        }
    }
}

// ---------------------------------------------------------------------------
// Fallback: fused dequant-in-loop kernel (used when ws can't hold operands)
// ---------------------------------------------------------------------------
__global__ __launch_bounds__(256)
void w4a32_wmma_gemm(const float* __restrict__ X,
                     const int*   __restrict__ Wp,
                     const float* __restrict__ scale,
                     const float* __restrict__ offs,
                     const float* __restrict__ bias,
                     float*       __restrict__ Y) {
    __shared__ uint16_t Als[BM][BK];
    __shared__ uint16_t Bls[BN][BK];

    const int tid  = threadIdx.x;
    const int wave = tid >> 5;
    const int lane = tid & 31;
    const int half = lane >> 4;
    const int lr   = lane & 15;

    const int m0 = blockIdx.y * BM;
    const int n0 = blockIdx.x * BN;
    const int wm = wave & 3;
    const int wn = wave >> 2;

    const int ra = tid >> 1;
    const int ca = (tid & 1) * 32;
    const int jb = tid & 63;
    const int kb = (tid >> 6) * 16;

    const float sc0 = scale[n0 + 2 * jb + 0];
    const float sc1 = scale[n0 + 2 * jb + 1];
    const float of0 = offs [n0 + 2 * jb + 0];
    const float of1 = offs [n0 + 2 * jb + 1];

    v8f acc[2][4];
    #pragma unroll
    for (int i = 0; i < 2; ++i)
        #pragma unroll
        for (int j = 0; j < 4; ++j)
            acc[i][j] = v8f{};

    const float* Aptr = X  + (size_t)(m0 + ra) * K + ca;
    const int*   Bptr = Wp + (size_t)kb * NPK + (n0 / 2 + jb);

    float4 aReg[8];
    int    bReg[16];
    #pragma unroll
    for (int i = 0; i < 8; ++i)  aReg[i] = ((const float4*)Aptr)[i];
    #pragma unroll
    for (int i = 0; i < 16; ++i) bReg[i] = Bptr[(size_t)i * NPK];

    for (int kt = 0; kt < K / BK; ++kt) {
        __syncthreads();
        uint32_t* arow = (uint32_t*)&Als[ra][ca];
        #pragma unroll
        for (int i = 0; i < 8; ++i) {
            float4 f = aReg[i];
            arow[2 * i + 0] = pack_bf16x2(f.x, f.y);
            arow[2 * i + 1] = pack_bf16x2(f.z, f.w);
        }
        uint32_t* b0 = (uint32_t*)&Bls[2 * jb + 0][kb];
        uint32_t* b1 = (uint32_t*)&Bls[2 * jb + 1][kb];
        #pragma unroll
        for (int i = 0; i < 8; ++i) {
            int p0 = bReg[2 * i + 0];
            int p1 = bReg[2 * i + 1];
            float l0 = (float)(((p0 & 15) ^ 8) - 8);
            float h0 = (float)((((p0 >> 4) & 15) ^ 8) - 8);
            float l1 = (float)(((p1 & 15) ^ 8) - 8);
            float h1 = (float)((((p1 >> 4) & 15) ^ 8) - 8);
            b0[i] = pack_bf16x2((l0 + of0) * sc0, (l1 + of0) * sc0);
            b1[i] = pack_bf16x2((h0 + of1) * sc1, (h1 + of1) * sc1);
        }
        __syncthreads();

        if (kt + 1 < K / BK) {
            const float* An = Aptr + (size_t)(kt + 1) * BK;
            const int*   Bn = Bptr + (size_t)(kt + 1) * BK * NPK;
            #pragma unroll
            for (int i = 0; i < 8; ++i)  aReg[i] = ((const float4*)An)[i];
            #pragma unroll
            for (int i = 0; i < 16; ++i) bReg[i] = Bn[(size_t)i * NPK];
        }

        #pragma unroll
        for (int ks = 0; ks < 2; ++ks) {
            const int k0 = ks * 32 + half * 8;
            FragAB a[2], b[4];
            #pragma unroll
            for (int mt = 0; mt < 2; ++mt) {
                const uint16_t* p = &Als[wm * 32 + mt * 16 + lr][k0];
                a[mt].h[0] = *(const u16x8*)(p);
                a[mt].h[1] = *(const u16x8*)(p + 16);
            }
            #pragma unroll
            for (int nt = 0; nt < 4; ++nt) {
                const uint16_t* p = &Bls[wn * 64 + nt * 16 + lr][k0];
                b[nt].h[0] = *(const u16x8*)(p);
                b[nt].h[1] = *(const u16x8*)(p + 16);
            }
            #pragma unroll
            for (int mt = 0; mt < 2; ++mt)
                #pragma unroll
                for (int nt = 0; nt < 4; ++nt)
                    acc[mt][nt] = __builtin_amdgcn_wmma_f32_16x16x32_bf16(
                        false, a[mt].v, false, b[nt].v,
                        (short)0, acc[mt][nt], false, false);
        }
    }

    #pragma unroll
    for (int nt = 0; nt < 4; ++nt) {
        const int col  = n0 + wn * 64 + nt * 16 + lr;
        const float bc = bias[col];
        #pragma unroll
        for (int mt = 0; mt < 2; ++mt) {
            const int rbase = m0 + wm * 32 + mt * 16 + half * 8;
            #pragma unroll
            for (int i = 0; i < 8; ++i)
                Y[(size_t)(rbase + i) * N + col] = acc[mt][nt][i] + bc;
        }
    }
}

// ---------------------------------------------------------------------------
extern "C" void kernel_launch(void* const* d_in, const int* in_sizes, int n_in,
                              void* d_out, int out_size, void* d_ws, size_t ws_size,
                              hipStream_t stream) {
    const float* x  = (const float*)d_in[0];
    const int*   wp = (const int*)  d_in[1];
    const float* sc = (const float*)d_in[2];
    const float* of = (const float*)d_in[3];
    const float* bs = (const float*)d_in[4];
    float*       y  = (float*)d_out;

    const size_t xbBytes = (size_t)M * K * sizeof(uint16_t);   // 67,108,864
    const size_t wtBytes = (size_t)N * K * sizeof(uint16_t);   // 90,177,536

    dim3 grid(N / BN, M / BM);   // 86 x 64

    if (ws_size >= xbBytes + wtBytes) {
        uint16_t* xb = (uint16_t*)d_ws;
        uint16_t* wt = (uint16_t*)((char*)d_ws + xbBytes);

        cvt_x_kernel<<<(M * K) / (256 * 8), 256, 0, stream>>>(x, xb);
        dequant_wt_kernel<<<dim3(NPK / 64, K / 64), 256, 0, stream>>>(wp, sc, of, wt);
        bf16_wmma_gemm<<<grid, 256, 0, stream>>>(xb, wt, bs, y);
    } else {
        w4a32_wmma_gemm<<<grid, 256, 0, stream>>>(x, wp, sc, of, bs, y);
    }
}
